// Relative_Temporal_SelfAttention_18133351923887
// MI455X (gfx1250) — matrix operand order repaired and checked by reading, not compiled, not run on hardware
//
#include <hip/hip_runtime.h>
#include <hip/hip_bf16.h>

// ---------------- problem constants ----------------
constexpr int B_  = 8;
constexpr int L_  = 1024;
constexpr int D_  = 512;
constexpr int H_  = 8;
constexpr int DH_ = 64;                 // head dim
constexpr float SCALE_ = 0.125f;        // 1/sqrt(64)

constexpr int RELP_ = 24;               // padded row stride (16 rows + 8 pad) for rel LDS

typedef __attribute__((ext_vector_type(16))) __bf16 v16bf;
typedef __attribute__((ext_vector_type(8)))  __bf16 v8bf;
typedef __attribute__((ext_vector_type(8)))  float  v8f;

// ---------------- WMMA fragment helpers (wave32, 16x16x32 bf16) ----------------
// A fragment 16x32 (MxK): lane = 16*kgroup + m ; elem i<8 -> K=kg*8+i ; i>=8 -> K=16+kg*8+(i-8)
__device__ __forceinline__ v16bf frag_a(const __bf16* __restrict__ p, int ld, int lane) {
  const int m  = lane & 15;
  const int kg = lane >> 4;
  const __bf16* base = p + (size_t)m * ld + kg * 8;
  v8bf lo = *(const v8bf*)(base);
  v8bf hi = *(const v8bf*)(base + 16);
  return __builtin_shufflevector(lo, hi, 0,1,2,3,4,5,6,7,8,9,10,11,12,13,14,15);
}

// B fragment 32x16 (KxN) read from *transposed* storage pT[n][k] (row n, contiguous k):
// lane = 16*kgroup + n ; elem i<8 -> K=kg*8+i ; i>=8 -> K=16+kg*8+(i-8)
__device__ __forceinline__ v16bf frag_b(const __bf16* __restrict__ p, int ld, int lane) {
  const int n  = lane & 15;
  const int kh = lane >> 4;
  const __bf16* base = p + (size_t)n * ld + kh * 8;
  v8bf lo = *(const v8bf*)(base);
  v8bf hi = *(const v8bf*)(base + 16);
  return __builtin_shufflevector(lo, hi, 0,1,2,3,4,5,6,7,8,9,10,11,12,13,14,15);
}

__device__ __forceinline__ v8f wmma_bf(v16bf a, v16bf b, v8f c) {
  return __builtin_amdgcn_wmma_f32_16x16x32_bf16(false, a, false, b, (short)0, c, false, false);
}

// ---------------- K0: conversions ----------------
__global__ void cvt_f32_bf16(const float* __restrict__ src, __bf16* __restrict__ dst, int n) {
  int i = blockIdx.x * blockDim.x + threadIdx.x;
  if (i < n) dst[i] = (__bf16)src[i];
}

// dst[c][r] = (bf16)src[r][c] ; src is rows x cols, row-major
__global__ void transpose_to_bf16(const float* __restrict__ src, __bf16* __restrict__ dst,
                                  int rows, int cols) {
  int c = blockIdx.x * blockDim.x + threadIdx.x;
  int r = blockIdx.y * blockDim.y + threadIdx.y;
  if (c < cols && r < rows) dst[(size_t)c * rows + r] = (__bf16)src[(size_t)r * cols + c];
}

// ---------------- K1: fused QKV projection GEMM ----------------
// out rows = B*L tokens, cols = D features. Each wave computes a 32x32 tile.
// q,k stored [B,H,L,DH]; v stored transposed [B,H,DH,L].
__global__ __launch_bounds__(128) void qkv_gemm(
    const __bf16* __restrict__ xb,
    const __bf16* __restrict__ wqt, const __bf16* __restrict__ wkt, const __bf16* __restrict__ wvt,
    __bf16* __restrict__ qb, __bf16* __restrict__ kb, __bf16* __restrict__ vt) {
  const int lane = threadIdx.x & 31;
  const int wv   = threadIdx.x >> 5;
  const int row0 = blockIdx.x * 64 + (wv >> 1) * 32;
  const int col0 = blockIdx.y * 64 + (wv & 1) * 32;
  const int wsel = blockIdx.z;
  const __bf16* wt = (wsel == 0) ? wqt : ((wsel == 1) ? wkt : wvt);

  v8f acc[2][2] = {};
  for (int k0 = 0; k0 < D_; k0 += 32) {
    v16bf a0 = frag_a(xb + (size_t)row0 * D_ + k0, D_, lane);
    v16bf a1 = frag_a(xb + (size_t)(row0 + 16) * D_ + k0, D_, lane);
    v16bf b0 = frag_b(wt + (size_t)col0 * D_ + k0, D_, lane);
    v16bf b1 = frag_b(wt + (size_t)(col0 + 16) * D_ + k0, D_, lane);
    acc[0][0] = wmma_bf(a0, b0, acc[0][0]);
    acc[0][1] = wmma_bf(a0, b1, acc[0][1]);
    acc[1][0] = wmma_bf(a1, b0, acc[1][0]);
    acc[1][1] = wmma_bf(a1, b1, acc[1][1]);
  }

  const int kg = lane >> 4;   // C layout: vgpr r holds row m = r + 8*kg, col n = lane&15
  const int nn = lane & 15;
#pragma unroll
  for (int i = 0; i < 2; ++i)
#pragma unroll
    for (int j = 0; j < 2; ++j)
#pragma unroll
      for (int r = 0; r < 8; ++r) {
        int rg = row0 + i * 16 + 8 * kg + r;     // token index
        int cg = col0 + j * 16 + nn;             // feature index
        int b  = rg / L_;
        int l  = rg - b * L_;
        int h  = cg / DH_;
        int d  = cg - h * DH_;
        float val = acc[i][j][r];
        if (wsel == 0)
          qb[(((size_t)b * H_ + h) * L_ + l) * DH_ + d] = (__bf16)val;
        else if (wsel == 1)
          kb[(((size_t)b * H_ + h) * L_ + l) * DH_ + d] = (__bf16)val;
        else
          vt[(((size_t)b * H_ + h) * DH_ + d) * L_ + l] = (__bf16)val;
      }
}

// ---------------- K2: fused flash attention with relative multiplicative bias ----------------
// One wave (32 threads) per workgroup; wave owns query rows [l0, l0+16).
// Phase 1: compute this wave's 16 rows of R = q @ We straight into LDS (128 WMMAs).
// Phase 2: flash loop over keys in 32-wide tiles, gathering rel = R[l, L-1-|s-l|] from LDS.
__global__ __launch_bounds__(32) void attn(
    const __bf16* __restrict__ qb, const __bf16* __restrict__ kb,
    const __bf16* __restrict__ vt, const __bf16* __restrict__ wet,
    float* __restrict__ out) {
  // rel stored column-major [e][RELP_]: store = 1x ds_store_b128/lane/tile (16B aligned),
  // gather = conflict-free (stride 24 elem -> bank step 12, 16 distinct banks).
  __shared__ __attribute__((aligned(16))) __bf16 relc[L_ * RELP_];   // 48 KB
  __shared__ __attribute__((aligned(16))) __bf16 plds[16 * 32];      //  1 KB

  const int lane = threadIdx.x & 31;
  const int bh   = blockIdx.y;
  const int b    = bh >> 3;
  const int h    = bh & 7;
  const int l0   = blockIdx.x * 16;

  const __bf16* qh = qb + (size_t)bh * L_ * DH_;
  const __bf16* kh = kb + (size_t)bh * L_ * DH_;
  const __bf16* vh = vt + (size_t)bh * DH_ * L_;

  const int kg = lane >> 4;
  const int nn = lane & 15;

  // q: A-fragments for K = 0..31 and 32..63 (held for whole kernel)
  v16bf qa0 = frag_a(qh + (size_t)l0 * DH_ + 0,  DH_, lane);
  v16bf qa1 = frag_a(qh + (size_t)l0 * DH_ + 32, DH_, lane);

  // ---- Phase 1: rel rows into LDS ----
  for (int e0 = 0; e0 < L_; e0 += 16) {
    v16bf wb0 = frag_b(wet + (size_t)e0 * DH_ + 0,  DH_, lane);
    v16bf wb1 = frag_b(wet + (size_t)e0 * DH_ + 32, DH_, lane);
    v8f c = {};
    c = wmma_bf(qa0, wb0, c);
    c = wmma_bf(qa1, wb1, c);
    v8bf pk;
#pragma unroll
    for (int i = 0; i < 8; ++i) pk[i] = (__bf16)c[i];
    // C elem [m][n]: m = 8*kg + r (8 consecutive rows per lane), n = nn -> column e0+nn
    *(v8bf*)(&relc[(e0 + nn) * RELP_ + kg * 8]) = pk;
  }
  // single-wave workgroup: same-wave LDS ops are in-order, no barrier needed.

  // ---- Phase 2: flash attention ----
  v8f o[4] = {};                 // 16 x 64 output accumulator (4 col tiles of dh)
  float rmax[8], rsum[8];
#pragma unroll
  for (int r = 0; r < 8; ++r) { rmax[r] = -1.0e30f; rsum[r] = 0.0f; }

  for (int s0 = 0; s0 < L_; s0 += 32) {
    if (s0 + 32 < L_) {
      __builtin_prefetch(kh + (size_t)(s0 + 32) * DH_, 0, 1);
      __builtin_prefetch(vh + (size_t)(s0 + 32), 0, 1);
    }

    // ---- S = q @ k^T  (two 16x16 column tiles) ----
    v8f Sc[2] = {};
#pragma unroll
    for (int j = 0; j < 2; ++j) {
      v16bf kb0 = frag_b(kh + (size_t)(s0 + j * 16) * DH_ + 0,  DH_, lane);
      v16bf kb1 = frag_b(kh + (size_t)(s0 + j * 16) * DH_ + 32, DH_, lane);
      Sc[j] = wmma_bf(qa0, kb0, Sc[j]);
      Sc[j] = wmma_bf(qa1, kb1, Sc[j]);
    }

    // ---- multiplicative relative bias + scale (rel gathered from LDS) ----
    float lg[2][8];
#pragma unroll
    for (int j = 0; j < 2; ++j)
#pragma unroll
      for (int r = 0; r < 8; ++r) {
        int lr   = 8 * kg + r;            // local row 0..15
        int lrow = l0 + lr;
        int scol = s0 + j * 16 + nn;
        int dd   = scol - lrow;
        int ad   = dd < 0 ? -dd : dd;
        float rel = (float)relc[(L_ - 1 - ad) * RELP_ + lr];
        lg[j][r] = SCALE_ * Sc[j][r] * rel;
      }

    // ---- online softmax (row = 16 lanes within a half-wave) ----
#pragma unroll
    for (int r = 0; r < 8; ++r) {
      float mx = fmaxf(lg[0][r], lg[1][r]);
#pragma unroll
      for (int off = 1; off < 16; off <<= 1) mx = fmaxf(mx, __shfl_xor(mx, off, 32));
      float mnew = fmaxf(rmax[r], mx);
      float corr = __expf(rmax[r] - mnew);
      rmax[r] = mnew;
      float p0 = __expf(lg[0][r] - mnew);
      float p1 = __expf(lg[1][r] - mnew);
      float ps = p0 + p1;
#pragma unroll
      for (int off = 1; off < 16; off <<= 1) ps += __shfl_xor(ps, off, 32);
      rsum[r] = rsum[r] * corr + ps;
#pragma unroll
      for (int t = 0; t < 4; ++t) o[t][r] *= corr;
      lg[0][r] = p0;
      lg[1][r] = p1;
    }

    // ---- relayout P: C-fragment -> LDS -> A-fragment ----
#pragma unroll
    for (int j = 0; j < 2; ++j)
#pragma unroll
      for (int r = 0; r < 8; ++r) {
        int m = 8 * kg + r;
        plds[m * 32 + j * 16 + nn] = (__bf16)lg[j][r];
      }
    v16bf pa = frag_a(plds, 32, lane);

    // ---- O += P @ V  (V pre-transposed: vh[d][s], contiguous s) ----
#pragma unroll
    for (int t = 0; t < 4; ++t) {
      v16bf vb = frag_b(vh + (size_t)(t * 16) * L_ + s0, L_, lane);
      o[t] = wmma_bf(pa, vb, o[t]);
    }
  }

  // ---- finalize: divide by row sum, write fp32 output [B, L, D] ----
#pragma unroll
  for (int t = 0; t < 4; ++t)
#pragma unroll
    for (int r = 0; r < 8; ++r) {
      int lrow = l0 + 8 * kg + r;
      int dcol = h * DH_ + t * 16 + nn;
      out[((size_t)b * L_ + lrow) * D_ + dcol] = o[t][r] / rsum[r];
    }
}

// ---------------- host launcher ----------------
extern "C" void kernel_launch(void* const* d_in, const int* in_sizes, int n_in,
                              void* d_out, int out_size, void* d_ws, size_t ws_size,
                              hipStream_t stream) {
  const float* x  = (const float*)d_in[0];
  const float* Wq = (const float*)d_in[1];
  const float* Wk = (const float*)d_in[2];
  const float* Wv = (const float*)d_in[3];
  const float* We = (const float*)d_in[4];
  float* out = (float*)d_out;

  constexpr size_t nX  = (size_t)B_ * L_ * D_;       // 4,194,304
  constexpr size_t nW  = (size_t)D_ * D_;            //   262,144
  constexpr size_t nWe = (size_t)DH_ * L_;           //    65,536
  constexpr size_t nQ  = (size_t)B_ * H_ * L_ * DH_; // 4,194,304

  char* p = (char*)d_ws;
  __bf16* xb  = (__bf16*)p; p += nX  * 2;
  __bf16* wqt = (__bf16*)p; p += nW  * 2;
  __bf16* wkt = (__bf16*)p; p += nW  * 2;
  __bf16* wvt = (__bf16*)p; p += nW  * 2;
  __bf16* wet = (__bf16*)p; p += nWe * 2;
  __bf16* qb  = (__bf16*)p; p += nQ  * 2;
  __bf16* kb  = (__bf16*)p; p += nQ  * 2;
  __bf16* vt  = (__bf16*)p; p += nQ  * 2;   // total ~35 MB

  // K0: conversions
  cvt_f32_bf16<<<(int)((nX + 255) / 256), 256, 0, stream>>>(x, xb, (int)nX);
  dim3 tb(16, 16);
  transpose_to_bf16<<<dim3(D_ / 16, D_ / 16), tb, 0, stream>>>(Wq, wqt, D_, D_);
  transpose_to_bf16<<<dim3(D_ / 16, D_ / 16), tb, 0, stream>>>(Wk, wkt, D_, D_);
  transpose_to_bf16<<<dim3(D_ / 16, D_ / 16), tb, 0, stream>>>(Wv, wvt, D_, D_);
  transpose_to_bf16<<<dim3(L_ / 16, DH_ / 16), tb, 0, stream>>>(We, wet, DH_, L_);

  // K1: QKV projections (z = 0:q, 1:k, 2:v)
  qkv_gemm<<<dim3((B_ * L_) / 64, D_ / 64, 3), 128, 0, stream>>>(
      xb, wqt, wkt, wvt, qb, kb, vt);

  // K2: fused rel-projection + flash attention (single-wave workgroups)
  attn<<<dim3(L_ / 16, B_ * H_), 32, 0, stream>>>(qb, kb, vt, wet, out);
}
